// CRF_30064771072911
// MI455X (gfx1250) — matrix operand order repaired and checked by reading, not compiled, not run on hardware
//
#include <hip/hip_runtime.h>

// CRF forward (log partition) for B=512, T=1024, K=128 on gfx1250.
// Recurrence per step rewritten as exp-space GEMM: val = log( e @ W^T ) + m + feat
// with W = exp(trans) held in registers as WMMA B-fragments (f16), e in LDS (f16),
// f32 accumulation via v_wmma_f32_16x16x32_f16. Raw v_log_f32/v_exp_f32 transcendentals
// (inputs provably outside the denormal fixup range), feats software-pipelined in regs.

typedef __attribute__((ext_vector_type(16))) _Float16 v16h;
typedef __attribute__((ext_vector_type(8)))  float    v8f;

#define B_SZ   512
#define T_SZ   1024
#define K_SZ   128
#define BM     16          // batch rows per workgroup
#define LDA    136         // padded LDS row stride in f16 (128 + 8) -> conflict-free b128 loads
#define PIPE   4           // feats software-pipeline depth (steps)

#define LN2f    0.69314718055994530942f
#define LOG2Ef  1.44269504088896340736f

// raw hardware transcendentals (v_log_f32 = log2, v_exp_f32 = exp2)
__device__ __forceinline__ float fast_ln(float x) {   // natural log, x >= 0, not denormal
    return __builtin_amdgcn_logf(x) * LN2f;           // log2(0) = -inf -> -inf
}
__device__ __forceinline__ float fast_exp(float x) {  // e^x, x <= smallish
    return __builtin_amdgcn_exp2f(x * LOG2Ef);        // exp2(-inf) = 0
}

// order-preserving float <-> uint encoding so row-max can use native ds_max_u32
__device__ __forceinline__ unsigned enc_f32(float x) {
    unsigned b = __float_as_uint(x);
    return ((int)b < 0) ? ~b : (b | 0x80000000u);
}
__device__ __forceinline__ float dec_f32(unsigned u) {
    unsigned b = ((int)u < 0) ? (u ^ 0x80000000u) : ~u;
    return __uint_as_float(b);
}

// butterfly max across the 16 lanes of each wave-half via ds_swizzle (group-of-32, xor mode)
#define SWIZ_MAX(v, OFF)                                                             \
    do {                                                                             \
        float _o = __int_as_float(                                                   \
            __builtin_amdgcn_ds_swizzle(__float_as_int(v), (OFF)));                  \
        (v) = fmaxf((v), _o);                                                        \
    } while (0)

__global__ __launch_bounds__(256, 1)
void CRF_30064771072911_kernel(const float* __restrict__ feats,
                               const float* __restrict__ trans,
                               float* __restrict__ out)
{
    __shared__ __align__(16) _Float16 Abuf[2][BM][LDA];   // e = exp(alpha - m), double-buffered
    __shared__ unsigned               Mbuf[2][BM];        // encoded per-row shift (max), dbl-buf

    const int tid  = threadIdx.x;
    const int lane = tid & 31;
    const int wv   = tid >> 5;        // wave id 0..7 -> 'to' tile
    const int l16  = lane & 15;
    const int half = lane >> 4;       // 0 / 1
    const int b0   = blockIdx.x * BM;
    const int to   = wv * 16 + l16;   // this lane's output column (N)

    // ---- Build B-operand fragments once: B[k][n] = exp(trans[n][k]), f16, kept in VGPRs.
    // B 32x16 frag, contiguous k as in A: lane n=N (halves split K), VGPR v holds K = 16*half + {2v,2v+1}.
    v16h bfrag[4];
    {
        const float* trow = trans + (size_t)to * K_SZ;
        #pragma unroll
        for (int kb = 0; kb < 4; ++kb) {
            const float* p = trow + kb * 32 + half * 16;   // 16 consecutive frm values
            v16h f;
            #pragma unroll
            for (int i = 0; i < 16; ++i)
                f[i] = (_Float16)fast_exp(p[i]);           // exp(MIN_VAL) -> 0
            bfrag[kb] = f;
        }
    }

    // ---- Init state: alpha0 = MIN_VAL except START(=K-1)=0  =>  e0 = onehot(K-1), m0 = 0.
    for (int i = tid; i < BM * K_SZ; i += 256) {
        int r = i >> 7, c = i & (K_SZ - 1);
        Abuf[0][r][c] = (_Float16)((c == K_SZ - 1) ? 1.0f : 0.0f);
    }
    if (tid < BM) { Mbuf[0][tid] = 0u; Mbuf[1][tid] = 0u; }
    __syncthreads();

    // ---- feats pointers + register pipeline (hide HBM latency behind PIPE steps)
    const float* fptr[8];
    float fq[PIPE][8];
    #pragma unroll
    for (int r = 0; r < 8; ++r) {
        const float* base =
            feats + (size_t)(b0 + r + half * 8) * (size_t)T_SZ * (size_t)K_SZ + to;
        #pragma unroll
        for (int d = 0; d < PIPE; ++d)
            fq[d][r] = __builtin_nontemporal_load(base + (size_t)d * K_SZ);
        fptr[r] = base + (size_t)PIPE * K_SZ;   // points at step t+PIPE
    }

    float mreg[8];
    #pragma unroll
    for (int r = 0; r < 8; ++r) mreg[r] = 0.0f;

    for (int tb = 0; tb < T_SZ; tb += PIPE) {
        #pragma unroll
        for (int u = 0; u < PIPE; ++u) {
            const int t   = tb + u;
            const int cur = t & 1, nxt = cur ^ 1;

            // consume pipelined feats for step t, then refill slot with step t+PIPE
            float fv[8];
            #pragma unroll
            for (int r = 0; r < 8; ++r) fv[r] = fq[u][r];
            if (t + PIPE < T_SZ) {
                #pragma unroll
                for (int r = 0; r < 8; ++r) {
                    fq[u][r] = __builtin_nontemporal_load(fptr[r]);
                    fptr[r] += K_SZ;
                }
            }

            // A-fragments (16x32 f16) from LDS: two contiguous 16B chunks per lane.
            // acc[b,to] = sum_frm e[b,frm] * W[to,frm]; two chains for WMMA ILP.
            v8f acc0 = {}, acc1 = {};
            #pragma unroll
            for (int kb = 0; kb < 4; ++kb) {
                const _Float16* basep = &Abuf[cur][l16][kb * 32 + half * 16];
                v16h a;
                *(uint4*)&a       = *(const uint4*)(basep);
                *((uint4*)&a + 1) = *(const uint4*)(basep + 8);
                if (kb & 1)
                    acc1 = __builtin_amdgcn_wmma_f32_16x16x32_f16(
                        false, a, false, bfrag[kb], (short)0, acc1, false, false);
                else
                    acc0 = __builtin_amdgcn_wmma_f32_16x16x32_f16(
                        false, a, false, bfrag[kb], (short)0, acc0, false, false);
            }

            // val = ln(acc) + m_old + feat   (D layout: VGPR r -> row r + half*8, N = lane)
            // acc is a sum of non-negative f16*f16 products: zero or >= ~2^-48, never
            // f32-denormal, so the raw v_log_f32 path is exact enough (log2(0) = -inf).
            float val[8];
            #pragma unroll
            for (int r = 0; r < 8; ++r) {
                float lg = __builtin_amdgcn_logf(acc0[r] + acc1[r]);   // log2
                val[r] = fmaf(lg, LN2f, mreg[r] + fv[r]);
            }

            // per-row max over this wave's 16 columns, then cross-wave via ds_max_u32
            float pmx[8];
            #pragma unroll
            for (int r = 0; r < 8; ++r) {
                pmx[r] = val[r];
                SWIZ_MAX(pmx[r], 0x041F);   // xor 1
                SWIZ_MAX(pmx[r], 0x081F);   // xor 2
                SWIZ_MAX(pmx[r], 0x101F);   // xor 4
                SWIZ_MAX(pmx[r], 0x201F);   // xor 8
            }
            if (l16 == 0) {                 // lanes 0 and 16 publish rows 0-7 / 8-15
                #pragma unroll
                for (int r = 0; r < 8; ++r)
                    atomicMax(&Mbuf[nxt][r + half * 8], enc_f32(pmx[r]));
            }
            __syncthreads();                // Mbuf[nxt] complete; Abuf[cur] fully consumed

            #pragma unroll
            for (int r = 0; r < 8; ++r)
                mreg[r] = dec_f32(Mbuf[nxt][r + half * 8]);
            if (tid < BM) Mbuf[cur][tid] = 0u;   // reset for step t+1's atomics

            // e_new = exp(val - m_new) in [0,1] -> f16 into Abuf[nxt]
            #pragma unroll
            for (int r = 0; r < 8; ++r) {
                float e = __builtin_amdgcn_exp2f((val[r] - mreg[r]) * LOG2Ef);
                Abuf[nxt][r + half * 8][to] = (_Float16)e;
            }
            __syncthreads();                // Abuf[nxt] + Mbuf reset visible
        }
    }

    // ---- Termination: logZ[b] = m + log( sum_to e[row][to] * exp(trans[END][to]) )
    // After t = T-1 (odd), state lives in Abuf[0] / Mbuf[0].
    if (tid < BM) {
        const int    row = tid;
        const float* te  = trans + (size_t)(K_SZ - 2) * K_SZ;   // END row
        float s = 0.0f;
        for (int c = 0; c < K_SZ; ++c)
            s += (float)Abuf[0][row][c] * fast_exp(te[c]);
        out[b0 + row] = dec_f32(Mbuf[0][row]) + fast_ln(s);
    }
}

extern "C" void kernel_launch(void* const* d_in, const int* in_sizes, int n_in,
                              void* d_out, int out_size, void* d_ws, size_t ws_size,
                              hipStream_t stream) {
    const float* feats = (const float*)d_in[0];   // [B, T, K] f32
    const float* trans = (const float*)d_in[1];   // [K, K]    f32
    float*       out   = (float*)d_out;           // [B]       f32
    (void)in_sizes; (void)n_in; (void)out_size; (void)d_ws; (void)ws_size;

    dim3 grid(B_SZ / BM);   // 32 workgroups, one independent 16-row recurrence each
    dim3 block(256);        // 8 wave32 waves -> 8 'to' tiles of 16
    CRF_30064771072911_kernel<<<grid, block, 0, stream>>>(feats, trans, out);
}
